// MultiHeadSelfAttention_71442486001955
// MI455X (gfx1250) — matrix operand order repaired and checked by reading, not compiled, not run on hardware
//
#include <hip/hip_runtime.h>

#define DM 512
#define NH 8
#define DH 64
#define BB 4
#define SS 2048
#define MTOT (BB * SS)          // 8192 rows of (b,s)
#define NEL ((size_t)MTOT * DM) // 4,194,304 elements per activation buffer
#define WEL ((size_t)DM * DM)   //   262,144 elements per weight matrix
#define KSTAGES (DM / 32)       // 16 pipeline stages of K=32

typedef __attribute__((ext_vector_type(16))) _Float16 v16h;
typedef __attribute__((ext_vector_type(8)))  _Float16 v8h;
typedef __attribute__((ext_vector_type(8)))  float    v8f;

static __device__ __forceinline__ v8h  ld_v8h (const _Float16* p) { return *(const v8h *)p; }
static __device__ __forceinline__ v16h ld_v16h(const _Float16* p) { return *(const v16h*)p; }

// ---------------------------------------------------------------------------
// Streaming fp32 -> f16 conversion (one pass; all GEMM hot loops then run on
// native f16 operands: pure b128 loads -> v_wmma, no cvt chains).
// ---------------------------------------------------------------------------
__global__ __launch_bounds__(256)
void cvt_f32_to_f16_kernel(const float* __restrict__ src,
                           _Float16* __restrict__ dst, int n8)
{
    int i = blockIdx.x * blockDim.x + threadIdx.x;
    if (i < n8) {
        v8f x = ((const v8f*)src)[i];
        v8h y;
#pragma unroll
        for (int j = 0; j < 8; ++j) y[j] = (_Float16)x[j];
        ((v8h*)dst)[i] = y;
    }
}

// ---------------------------------------------------------------------------
// One K=32 stage of fragment loads for X @ W^T (wave tile 16x64).
// A-frag: lane = row (lane&15), halves K = lhalf*16+[0..15] (two 8-wide loads).
// B-frag: lane = col (lane&15), halves K = lhalf*16+[0..15] (contig in W row).
// ---------------------------------------------------------------------------
static __device__ __forceinline__ void load_stage(
    const _Float16* __restrict__ arowp, const _Float16* __restrict__ W,
    int colbase, int lhalf, int lmod, int kk, v16h& a, v16h (&bw)[4])
{
    const int kb = kk + lhalf * 16;
    v8h x0 = ld_v8h(arowp + kb);
    v8h x1 = ld_v8h(arowp + kb + 8);
#pragma unroll
    for (int i = 0; i < 8; ++i) { a[i] = x0[i]; a[8 + i] = x1[i]; }
    const int kb2 = kk + lhalf * 16;
#pragma unroll
    for (int nc = 0; nc < 4; ++nc) {
        const int n = colbase + nc * 16 + lmod;
        bw[nc] = ld_v16h(W + (size_t)n * DM + kb2);
    }
}

// ---------------------------------------------------------------------------
// QKV projection: out = X @ W^T + bias (f16 operands, f32 accumulate).
// mode 0: out[b][h][s][d]   (Q, K)
// mode 1: out[b][h][d][s]   (V transposed so PV B-frags are contiguous)
// Software-pipelined: stage i+1 fragment loads issued before stage i WMMAs;
// a compiler fence keeps the order, so waits become partial and loads overlap
// the matrix ops.
// ---------------------------------------------------------------------------
__global__ __launch_bounds__(256)
void qkv_proj_kernel(const _Float16* __restrict__ X, const _Float16* __restrict__ W,
                     const float* __restrict__ bias, _Float16* __restrict__ out,
                     int mode)
{
    const int lane   = threadIdx.x & 31;
    const int w      = threadIdx.x >> 5;
    const int lhalf  = lane >> 4;   // 0/1
    const int lmod   = lane & 15;
    const int rowbase = blockIdx.y * 128 + w * 16;
    const int colbase = blockIdx.x * 64;

    v8f acc[4] = {};
    const _Float16* arowp = X + (size_t)(rowbase + lmod) * DM;

    v16h aF[2];
    v16h bF[2][4];
    load_stage(arowp, W, colbase, lhalf, lmod, 0, aF[0], bF[0]);

#pragma unroll
    for (int it = 0; it < KSTAGES; ++it) {
        const int cb = it & 1, nb = cb ^ 1;
        if (it + 1 < KSTAGES)
            load_stage(arowp, W, colbase, lhalf, lmod, (it + 1) * 32, aF[nb], bF[nb]);
        // fence: next-stage loads stay above this stage's WMMAs
        asm volatile("" ::: "memory");
#pragma unroll
        for (int nc = 0; nc < 4; ++nc)
            acc[nc] = __builtin_amdgcn_wmma_f32_16x16x32_f16(
                false, aF[cb], false, bF[cb][nc], (short)0, acc[nc], false, false);
    }

    // D layout: lane holds col (lane&15), rows (lane>>4)*8 + r
#pragma unroll
    for (int nc = 0; nc < 4; ++nc) {
#pragma unroll
        for (int r = 0; r < 8; ++r) {
            const int m   = rowbase + lhalf * 8 + r;
            const int col = colbase + nc * 16 + lmod;
            const float val = acc[nc][r] + bias[col];
            const int b = m / SS, s = m % SS;
            const int h = col / DH, d = col % DH;
            const size_t idx = (mode == 0)
                ? ((((size_t)b * NH + h) * SS + s) * DH + d)
                : ((((size_t)b * NH + h) * DH + d) * SS + s);
            out[idx] = (_Float16)val;
        }
    }
}

// ---------------------------------------------------------------------------
// Flash attention: one wave per 16-row query tile, online softmax over keys
// in blocks of 32. Scores & P never leave the WGP. V-frag loads are hoisted
// above the softmax so they hide behind ~100 VALU/trans ops.
// Grid: (S/128, B*H); block 256 = 8 waves.
// ---------------------------------------------------------------------------
__global__ __launch_bounds__(256)
void flash_attn_kernel(const _Float16* __restrict__ Qh,
                       const _Float16* __restrict__ Kh,
                       const _Float16* __restrict__ Vt,
                       _Float16* __restrict__ Ao)
{
    __shared__ __align__(32) _Float16 p_lds[8 * 16 * 32]; // 8 KB, wave-private tiles

    const int lane  = threadIdx.x & 31;
    const int w     = threadIdx.x >> 5;
    const int lhalf = lane >> 4;
    const int lmod  = lane & 15;

    const int bh = blockIdx.y;
    const int b  = bh / NH;
    const int h  = bh % NH;
    const int qbase = blockIdx.x * 128 + w * 16;

    const _Float16* Qb = Qh + (size_t)bh * SS * DH;
    const _Float16* Kb = Kh + (size_t)bh * SS * DH;
    const _Float16* Vb = Vt + (size_t)bh * DH * SS;
    _Float16* pw = p_lds + w * (16 * 32);

    // fold 1/sqrt(Dh) and log2(e) into Q so scores live in the exp2 domain
    const float qscale = 0.125f * 1.4426950408889634f;

    // Q A-frags for K(dim)=0..31 and 32..63, resident for whole kernel
    v16h qa[2];
    {
        const int qrow = qbase + lmod;
#pragma unroll
        for (int t = 0; t < 2; ++t) {
            const int kb = t * 32 + lhalf * 16;
            v8h x0 = ld_v8h(Qb + (size_t)qrow * DH + kb);
            v8h x1 = ld_v8h(Qb + (size_t)qrow * DH + kb + 8);
#pragma unroll
            for (int i = 0; i < 8; ++i) {
                qa[t][i]     = (_Float16)((float)x0[i] * qscale);
                qa[t][8 + i] = (_Float16)((float)x1[i] * qscale);
            }
        }
    }

    float mrow[8], lrow[8];
#pragma unroll
    for (int r = 0; r < 8; ++r) { mrow[r] = -1e30f; lrow[r] = 0.0f; }
    v8f o[4] = {};

    for (int kb = 0; kb < SS; kb += 32) {
        // ---- K frags + V frags issued together; V hides behind softmax ----
        v16h kf[2][2];
#pragma unroll
        for (int t = 0; t < 2; ++t) {
            const int kd = t * 32 + lhalf * 16;
            kf[t][0] = ld_v16h(Kb + (size_t)(kb + lmod) * DH + kd);
            kf[t][1] = ld_v16h(Kb + (size_t)(kb + 16 + lmod) * DH + kd);
        }
        v16h vfr[4];
#pragma unroll
        for (int nc = 0; nc < 4; ++nc) {
            const int d = nc * 16 + lmod;
            vfr[nc] = ld_v16h(Vb + (size_t)d * SS + kb + lhalf * 16);
        }
        if (kb + 32 < SS) {
            __builtin_prefetch(Kb + (size_t)(kb + 32 + lmod) * DH, 0, 1);
            __builtin_prefetch(Vb + (size_t)lmod * SS + kb + 32, 0, 1);
        }

        // ---- scores: c0 = keys kb..kb+15, c1 = keys kb+16..kb+31 ----
        v8f c0 = {}, c1 = {};
#pragma unroll
        for (int t = 0; t < 2; ++t) {
            c0 = __builtin_amdgcn_wmma_f32_16x16x32_f16(
                false, qa[t], false, kf[t][0], (short)0, c0, false, false);
            c1 = __builtin_amdgcn_wmma_f32_16x16x32_f16(
                false, qa[t], false, kf[t][1], (short)0, c1, false, false);
        }

        // ---- online softmax (log2 domain); rows m = lhalf*8 + r ----
        float alpha[8];
#pragma unroll
        for (int r = 0; r < 8; ++r) {
            const float s0 = c0[r], s1 = c1[r];
            float mx = fmaxf(s0, s1);
#pragma unroll
            for (int off = 1; off < 16; off <<= 1)
                mx = fmaxf(mx, __shfl_xor(mx, off, 32));
            const float mnew = fmaxf(mrow[r], mx);
            const float al = __builtin_amdgcn_exp2f(mrow[r] - mnew);
            const float p0 = __builtin_amdgcn_exp2f(s0 - mnew);
            const float p1 = __builtin_amdgcn_exp2f(s1 - mnew);
            float rs = p0 + p1;
#pragma unroll
            for (int off = 1; off < 16; off <<= 1)
                rs += __shfl_xor(rs, off, 32);
            lrow[r] = lrow[r] * al + rs;
            mrow[r] = mnew;
            alpha[r] = al;
            // stage P (D-layout) into LDS as a row-major 16x32 f16 tile
            const int m = lhalf * 8 + r;
            pw[m * 32 + lmod]      = (_Float16)p0;
            pw[m * 32 + 16 + lmod] = (_Float16)p1;
        }
#pragma unroll
        for (int nc = 0; nc < 4; ++nc)
#pragma unroll
            for (int r = 0; r < 8; ++r) o[nc][r] *= alpha[r];

        // wave-private LDS tile: drain DS stores before A-frag reload
        asm volatile("s_wait_dscnt 0" ::: "memory");

        // ---- reload P as an A-frag (16x32) ----
        v16h pa;
        {
            const int prow = lmod;
            const int kbl  = lhalf * 16;
            v8h x0 = ld_v8h(pw + prow * 32 + kbl);
            v8h x1 = ld_v8h(pw + prow * 32 + kbl + 8);
#pragma unroll
            for (int i = 0; i < 8; ++i) { pa[i] = x0[i]; pa[8 + i] = x1[i]; }
        }

        // ---- O += P @ V (V frags already in flight/registers) ----
#pragma unroll
        for (int nc = 0; nc < 4; ++nc)
            o[nc] = __builtin_amdgcn_wmma_f32_16x16x32_f16(
                false, pa, false, vfr[nc], (short)0, o[nc], false, false);
    }

    // ---- finalize: O /= l, write f16 [B,S,H*Dh] ----
#pragma unroll
    for (int r = 0; r < 8; ++r) {
        const float inv = 1.0f / lrow[r];
        const int s = qbase + lhalf * 8 + r;
#pragma unroll
        for (int nc = 0; nc < 4; ++nc) {
            const int col = h * DH + nc * 16 + lmod;
            Ao[((size_t)b * SS + s) * DM + col] = (_Float16)(o[nc][r] * inv);
        }
    }
}

// ---------------------------------------------------------------------------
// Output projection: Y = Ao @ Wo^T + bo (fp32 out, f16 operands),
// same software pipeline as the QKV projection.
// ---------------------------------------------------------------------------
__global__ __launch_bounds__(256)
void out_proj_kernel(const _Float16* __restrict__ Ao, const _Float16* __restrict__ Wo,
                     const float* __restrict__ bo, float* __restrict__ Y)
{
    const int lane   = threadIdx.x & 31;
    const int w      = threadIdx.x >> 5;
    const int lhalf  = lane >> 4;
    const int lmod   = lane & 15;
    const int rowbase = blockIdx.y * 128 + w * 16;
    const int colbase = blockIdx.x * 64;

    v8f acc[4] = {};
    const _Float16* arowp = Ao + (size_t)(rowbase + lmod) * DM;

    v16h aF[2];
    v16h bF[2][4];
    load_stage(arowp, Wo, colbase, lhalf, lmod, 0, aF[0], bF[0]);

#pragma unroll
    for (int it = 0; it < KSTAGES; ++it) {
        const int cb = it & 1, nb = cb ^ 1;
        if (it + 1 < KSTAGES)
            load_stage(arowp, Wo, colbase, lhalf, lmod, (it + 1) * 32, aF[nb], bF[nb]);
        asm volatile("" ::: "memory");
#pragma unroll
        for (int nc = 0; nc < 4; ++nc)
            acc[nc] = __builtin_amdgcn_wmma_f32_16x16x32_f16(
                false, aF[cb], false, bF[cb][nc], (short)0, acc[nc], false, false);
    }

#pragma unroll
    for (int nc = 0; nc < 4; ++nc) {
#pragma unroll
        for (int r = 0; r < 8; ++r) {
            const int m   = rowbase + lhalf * 8 + r;
            const int col = colbase + nc * 16 + lmod;
            Y[(size_t)m * DM + col] = acc[nc][r] + bo[col];
        }
    }
}

// ---------------------------------------------------------------------------
extern "C" void kernel_launch(void* const* d_in, const int* in_sizes, int n_in,
                              void* d_out, int out_size, void* d_ws, size_t ws_size,
                              hipStream_t stream)
{
    (void)in_sizes; (void)n_in; (void)out_size; (void)ws_size;

    const float* query = (const float*)d_in[0];
    const float* key   = (const float*)d_in[1];
    const float* value = (const float*)d_in[2];
    const float* Wq    = (const float*)d_in[3];
    const float* bq    = (const float*)d_in[4];
    const float* Wk    = (const float*)d_in[5];
    const float* bk    = (const float*)d_in[6];
    const float* Wv    = (const float*)d_in[7];
    const float* bv    = (const float*)d_in[8];
    const float* Wo    = (const float*)d_in[9];
    const float* bo    = (const float*)d_in[10];
    float* Y = (float*)d_out;

    // ---- workspace layout (f16 elements) ----
    _Float16* xq  = (_Float16*)d_ws;     // converted activations
    _Float16* xk  = xq  + NEL;
    _Float16* xv  = xk  + NEL;
    _Float16* wqh = xv  + NEL;           // converted weights
    _Float16* wkh = wqh + WEL;
    _Float16* wvh = wkh + WEL;
    _Float16* woh = wvh + WEL;
    _Float16* qh  = woh + WEL;           // projected, head-split
    _Float16* kh  = qh  + NEL;
    _Float16* vt  = kh  + NEL;           // V transposed per head
    _Float16* ao  = vt  + NEL;           // attention output

    dim3 blk(256, 1, 1);

    // ---- one-pass fp32 -> f16 conversions ----
    const int actN8 = (int)(NEL / 8);
    const int wN8   = (int)(WEL / 8);
    dim3 gAct((actN8 + 255) / 256, 1, 1);
    dim3 gW  ((wN8   + 255) / 256, 1, 1);
    cvt_f32_to_f16_kernel<<<gAct, blk, 0, stream>>>(query, xq, actN8);
    cvt_f32_to_f16_kernel<<<gAct, blk, 0, stream>>>(key,   xk, actN8);
    cvt_f32_to_f16_kernel<<<gAct, blk, 0, stream>>>(value, xv, actN8);
    cvt_f32_to_f16_kernel<<<gW,   blk, 0, stream>>>(Wq, wqh, wN8);
    cvt_f32_to_f16_kernel<<<gW,   blk, 0, stream>>>(Wk, wkh, wN8);
    cvt_f32_to_f16_kernel<<<gW,   blk, 0, stream>>>(Wv, wvh, wN8);
    cvt_f32_to_f16_kernel<<<gW,   blk, 0, stream>>>(Wo, woh, wN8);

    // ---- projections ----
    dim3 gProj(DM / 64, MTOT / 128, 1);   // (8, 64)
    qkv_proj_kernel<<<gProj, blk, 0, stream>>>(xq, wqh, bq, qh, 0);
    qkv_proj_kernel<<<gProj, blk, 0, stream>>>(xk, wkh, bk, kh, 0);
    qkv_proj_kernel<<<gProj, blk, 0, stream>>>(xv, wvh, bv, vt, 1);

    // ---- attention ----
    dim3 gAttn(SS / 128, BB * NH, 1);     // (16, 32)
    flash_attn_kernel<<<gAttn, blk, 0, stream>>>(qh, kh, vt, ao);

    // ---- output projection ----
    out_proj_kernel<<<gProj, blk, 0, stream>>>(ao, woh, bo, Y);
}
